// CustomRNN_86629490360878
// MI455X (gfx1250) — matrix operand order repaired and checked by reading, not compiled, not run on hardware
//
#include <hip/hip_runtime.h>
#include <hip/hip_bf16.h>
#include <math.h>

typedef __attribute__((ext_vector_type(16))) _Float16 v16h;
typedef __attribute__((ext_vector_type(8)))  _Float16 v8h;
typedef __attribute__((ext_vector_type(8)))  float    v8f;

#define IN_DIM   256
#define HID      512
#define OUT_DIM  64
#define BATCH_N  128
#define SEQ_L    1024

// ---------------------------------------------------------------------------
// Weight packing: f32 [K,N] row-major -> f16 tiles in WMMA B-operand layout.
// Tile (kt, nt): n = nt*16 + (L&15), k = kt*32 + (L<16 ? e : 16+e); each
// lane's 16 halves are contiguous in K and contiguous in memory (32 B/lane).
// ---------------------------------------------------------------------------
__global__ void pack_b_f16(const float* __restrict__ W, _Float16* __restrict__ dst,
                           int K, int N) {
    int total = K * N;
    for (int idx = blockIdx.x * blockDim.x + threadIdx.x; idx < total;
         idx += gridDim.x * blockDim.x) {
        int e    = idx & 15;
        int lane = (idx >> 4) & 31;
        int tile = idx >> 9;                 // 512 halves per tile
        int NT   = N >> 4;
        int kt   = tile / NT;
        int nt   = tile - kt * NT;
        int n = nt * 16 + (lane & 15);
        int k = kt * 32 + ((lane < 16) ? e : 16 + e);
        dst[idx] = (_Float16)W[(size_t)k * N + n];
    }
}

// 16x32 f16 A-tile from LDS in WMMA A layout (two ds_load_b128 per lane).
__device__ __forceinline__ v16h load_a_lds(const _Float16* base, int kt, int rs) {
    int lane = threadIdx.x & 31;
    int row  = lane & 15;
    int k0   = kt * 32 + ((lane < 16) ? 0 : 16);
    v8h a0 = *(const v8h*)(base + row * rs + k0);
    v8h a1 = *(const v8h*)(base + row * rs + k0 + 8);
    v16h r;
#pragma unroll
    for (int i = 0; i < 8; ++i) { r[i] = a0[i]; r[8 + i] = a1[i]; }
    return r;
}

// Packed 32x16 f16 B-tile (two global_load_b128 per lane, contiguous).
__device__ __forceinline__ v16h load_b_packed(const _Float16* p) {
    const v8h* q = (const v8h*)p;
    v8h b0 = q[0], b1 = q[1];
    v16h r;
#pragma unroll
    for (int i = 0; i < 8; ++i) { r[i] = b0[i]; r[8 + i] = b1[i]; }
    return r;
}

// Saturating fast tanh: 1 - 2/(exp2(2*log2e*x)+1); exact at +/-inf, ~f16-accurate.
__device__ __forceinline__ float fast_tanh(float x) {
    float z = __builtin_amdgcn_exp2f(x * 2.885390082f);
    return 1.0f - 2.0f * __builtin_amdgcn_rcpf(z + 1.0f);
}

// ---------------------------------------------------------------------------
// Persistent RNN scan: one workgroup (8 wave32) per 16 batch rows, whole
// 1024-step recurrence in-kernel, ONE barrier per step, x pipelined in regs.
// ---------------------------------------------------------------------------
__global__ __launch_bounds__(256, 1)
void rnn_scan_kernel(const float* __restrict__ x,        // [B, S, IN]
                     const _Float16* __restrict__ pWih,  // packed tiles [8][32]
                     const _Float16* __restrict__ pWhh,  // packed tiles [16][32]
                     const _Float16* __restrict__ pWho,  // packed tiles [16][4]
                     const float* __restrict__ b_hh,     // [HID]
                     const float* __restrict__ b_ho,     // [OUT]
                     float* __restrict__ out) {          // [B, S, OUT]
    __shared__ _Float16 lds_x[2][16 * IN_DIM];     // 2 x 8 KB ping-pong x_t
    __shared__ _Float16 lds_h[2][16 * HID];        // 2 x 16 KB ping-pong h

    const int tid  = threadIdx.x;
    const int lane = tid & 31;
    const int wave = tid >> 5;
    const int bb   = blockIdx.x * 16;              // batch base row
    const int NT_H = HID / 16;                     // 32 N-tiles over hidden dim
    const int nt0  = wave * 4;                     // this wave's 4 N-tiles

    // x register pipeline: each thread owns 16 contiguous floats of the tile.
    const int xr_row = tid >> 4;                   // 0..15
    const int xr_col = (tid & 15) * 16;            // 0..240
    float4 xr[4];

    // ---- prologue: h0 = 0, stage x(0) ----
    for (int i = tid; i < 16 * HID; i += 256) lds_h[0][i] = (_Float16)0.f;
    {
        const float4* p = (const float4*)&x[((size_t)(bb + xr_row) * SEQ_L + 0) * IN_DIM + xr_col];
#pragma unroll
        for (int i = 0; i < 4; ++i) xr[i] = p[i];
        _Float16* d = &lds_x[0][xr_row * IN_DIM + xr_col];
        v8h h0v, h1v;
#pragma unroll
        for (int i = 0; i < 2; ++i) {
            h0v[4*i+0] = (_Float16)xr[i].x;  h0v[4*i+1] = (_Float16)xr[i].y;
            h0v[4*i+2] = (_Float16)xr[i].z;  h0v[4*i+3] = (_Float16)xr[i].w;
            h1v[4*i+0] = (_Float16)xr[2+i].x; h1v[4*i+1] = (_Float16)xr[2+i].y;
            h1v[4*i+2] = (_Float16)xr[2+i].z; h1v[4*i+3] = (_Float16)xr[2+i].w;
        }
        ((v8h*)d)[0] = h0v; ((v8h*)d)[1] = h1v;
    }

    float bias_h[4];
#pragma unroll
    for (int i = 0; i < 4; ++i) bias_h[i] = b_hh[(nt0 + i) * 16 + (lane & 15)];
    const float bias_o = (wave < 4) ? b_ho[wave * 16 + (lane & 15)] : 0.f;

    __syncthreads();

    int cur = 0;
    for (int t = 0; t < SEQ_L; ++t) {
        const int nxt = cur ^ 1;

        // Issue global loads for x(t+1) early; waits will land after the GEMMs.
        {
            int tn = (t + 1 < SEQ_L) ? t + 1 : t;
            const float4* p = (const float4*)&x[((size_t)(bb + xr_row) * SEQ_L + tn) * IN_DIM + xr_col];
#pragma unroll
            for (int i = 0; i < 4; ++i) xr[i] = p[i];
        }

        v8f acc[4] = {};

        // GEMM1: acc += x_t(16x256) * W_ih(256x512)
        for (int kt = 0; kt < IN_DIM / 32; ++kt) {
            v16h a = load_a_lds(&lds_x[cur][0], kt, IN_DIM);
#pragma unroll
            for (int i = 0; i < 4; ++i) {
                v16h b = load_b_packed(pWih + (((size_t)kt * NT_H + nt0 + i) * 512 + lane * 16));
                acc[i] = __builtin_amdgcn_wmma_f32_16x16x32_f16(
                    false, a, false, b, (short)0, acc[i], false, false);
            }
        }
        // GEMM2: acc += h(16x512) * W_hh(512x512)
        for (int kt = 0; kt < HID / 32; ++kt) {
            v16h a = load_a_lds(&lds_h[cur][0], kt, HID);
#pragma unroll
            for (int i = 0; i < 4; ++i) {
                v16h b = load_b_packed(pWhh + (((size_t)kt * NT_H + nt0 + i) * 512 + lane * 16));
                acc[i] = __builtin_amdgcn_wmma_f32_16x16x32_f16(
                    false, a, false, b, (short)0, acc[i], false, false);
            }
        }

        // h_new = tanh(acc + b_hh) -> fp16 into the other h buffer.
#pragma unroll
        for (int i = 0; i < 4; ++i) {
            int n = (nt0 + i) * 16 + (lane & 15);
#pragma unroll
            for (int v = 0; v < 8; ++v) {
                int m = (lane < 16) ? v : 8 + v;   // C/D layout: VGPR v -> M=v / M=8+v
                lds_h[nxt][m * HID + n] = (_Float16)fast_tanh(acc[i][v] + bias_h[i]);
            }
        }

        // Commit x(t+1) registers into the other x buffer (packed stores).
        {
            _Float16* d = &lds_x[nxt][xr_row * IN_DIM + xr_col];
            v8h h0v, h1v;
#pragma unroll
            for (int i = 0; i < 2; ++i) {
                h0v[4*i+0] = (_Float16)xr[i].x;  h0v[4*i+1] = (_Float16)xr[i].y;
                h0v[4*i+2] = (_Float16)xr[i].z;  h0v[4*i+3] = (_Float16)xr[i].w;
                h1v[4*i+0] = (_Float16)xr[2+i].x; h1v[4*i+1] = (_Float16)xr[2+i].y;
                h1v[4*i+2] = (_Float16)xr[2+i].z; h1v[4*i+3] = (_Float16)xr[2+i].w;
            }
            ((v8h*)d)[0] = h0v; ((v8h*)d)[1] = h1v;
        }

        __syncthreads();   // the ONLY barrier per step

        // GEMM3: y_t(16x64) = h_new * W_ho + b_ho; waves 0..3 (one N-tile each),
        // waves 4..7 run ahead into next step's GEMM work.
        if (wave < 4) {
            v8f ya = {}, yb = {};
            for (int kt = 0; kt < HID / 32; kt += 2) {
                v16h a0 = load_a_lds(&lds_h[nxt][0], kt, HID);
                v16h b0 = load_b_packed(pWho + (((size_t)kt * (OUT_DIM / 16) + wave) * 512 + lane * 16));
                ya = __builtin_amdgcn_wmma_f32_16x16x32_f16(false, a0, false, b0, (short)0, ya, false, false);
                v16h a1 = load_a_lds(&lds_h[nxt][0], kt + 1, HID);
                v16h b1 = load_b_packed(pWho + (((size_t)(kt + 1) * (OUT_DIM / 16) + wave) * 512 + lane * 16));
                yb = __builtin_amdgcn_wmma_f32_16x16x32_f16(false, a1, false, b1, (short)0, yb, false, false);
            }
            int n = wave * 16 + (lane & 15);
#pragma unroll
            for (int v = 0; v < 8; ++v) {
                int m = (lane < 16) ? v : 8 + v;
                out[((size_t)(bb + m) * SEQ_L + t) * OUT_DIM + n] = ya[v] + yb[v] + bias_o;
            }
        }
        cur = nxt;
    }
}

// ---------------------------------------------------------------------------
extern "C" void kernel_launch(void* const* d_in, const int* in_sizes, int n_in,
                              void* d_out, int out_size, void* d_ws, size_t ws_size,
                              hipStream_t stream) {
    (void)in_sizes; (void)n_in; (void)out_size; (void)ws_size;
    const float* x    = (const float*)d_in[0];   // [128,1024,256]
    const float* W_hh = (const float*)d_in[1];   // [512,512]
    const float* W_ih = (const float*)d_in[2];   // [256,512]
    const float* b_hh = (const float*)d_in[3];   // [512]
    const float* W_ho = (const float*)d_in[4];   // [512,64]
    const float* b_ho = (const float*)d_in[5];   // [64]
    float* out = (float*)d_out;

    _Float16* ws   = (_Float16*)d_ws;
    _Float16* pWih = ws;                                   // 256*512 halves
    _Float16* pWhh = ws + (size_t)IN_DIM * HID;            // 512*512 halves
    _Float16* pWho = pWhh + (size_t)HID * HID;             // 512*64  halves

    pack_b_f16<<<(IN_DIM * HID + 255) / 256, 256, 0, stream>>>(W_ih, pWih, IN_DIM, HID);
    pack_b_f16<<<(HID * HID + 255) / 256, 256, 0, stream>>>(W_hh, pWhh, HID, HID);
    pack_b_f16<<<(HID * OUT_DIM + 255) / 256, 256, 0, stream>>>(W_ho, pWho, HID, OUT_DIM);

    rnn_scan_kernel<<<BATCH_N / 16, 256, 0, stream>>>(x, pWih, pWhh, pWho, b_hh, b_ho, out);
}
